// BiRnnAttentionCrfModel_32847909879872
// MI455X (gfx1250) — matrix operand order repaired and checked by reading, not compiled, not run on hardware
//
#include <hip/hip_runtime.h>
#include <hip/hip_bf16.h>
#include <math.h>

// Problem constants (from reference): B=64, T=512, D=256, V=32000, N=48
#define B_  64
#define T_  512
#define BT_ 32768            // B*T
#define D_  256
#define D2_ 512              // 2*D
#define G_  768              // 3*D (gates)
#define N_  48

typedef _Float16 f16;
typedef __attribute__((ext_vector_type(16))) _Float16 v16h;
typedef __attribute__((ext_vector_type(8)))  _Float16 v8h;
typedef __attribute__((ext_vector_type(8)))  float    v8f;
typedef __attribute__((ext_vector_type(4)))  unsigned int v4u;
typedef __attribute__((ext_vector_type(8)))  int v8i;
typedef __attribute__((ext_vector_type(4)))  int v4i;

// ---------------------------------------------------------------------------
// WMMA fragment helpers for v_wmma_f32_16x16x32_f16 (wave32).
// A (16x32 f16, row-major source, leading dim lda):
//   lane<16 : row=lane,    halves 0..7 -> K=k0..k0+7,   halves 8..15 -> K=k0+16..k0+23
//   lane>=16: row=lane-16, halves 0..7 -> K=k0+8..k0+15,halves 8..15 -> K=k0+24..k0+31
// => per lane: two contiguous 16-byte chunks.
// ---------------------------------------------------------------------------
__device__ __forceinline__ v16h load_a_frag(const f16* A, int lda, int m0, int k0, int lane) {
  int row = m0 + (lane & 15);
  int kb  = k0 + ((lane & 16) ? 8 : 0);
  const f16* p = A + (size_t)row * lda + kb;
  v8h c0 = *(const v8h*)p;
  v8h c1 = *(const v8h*)(p + 16);
  v16h r;
#pragma unroll
  for (int j = 0; j < 8; ++j) { r[j] = c0[j]; r[j + 8] = c1[j]; }
  return r;
}

// B (32x16 f16) from weights stored TRANSPOSED as [N][K] row-major:
//   col = n0 + (lane&15); same K split as A: lane<16: K=k0..k0+7,k0+16..k0+23 ; lane>=16: K=k0+8..k0+15,k0+24..k0+31
__device__ __forceinline__ v16h load_b_frag(const f16* Wt, int ldb, int n0, int k0, int lane) {
  int col = n0 + (lane & 15);
  int kb  = k0 + ((lane & 16) ? 8 : 0);
  const f16* p = Wt + (size_t)col * ldb + kb;
  v8h c0 = *(const v8h*)p;
  v8h c1 = *(const v8h*)(p + 16);
  v16h r;
#pragma unroll
  for (int j = 0; j < 8; ++j) { r[j] = c0[j]; r[j + 8] = c1[j]; }
  return r;
}

__device__ __forceinline__ v8f wmma_f16(v16h a, v16h b, v8f c) {
  return __builtin_amdgcn_wmma_f32_16x16x32_f16(false, a, false, b, (short)0, c, false, false);
}

// ---------------------------------------------------------------------------
// CDNA5 async global->LDS copy (ASYNCcnt path). LDS address = low 32 bits of
// the generic pointer (ISA 10.2: LDS aperture address truncates to addr[31:0]).
// ---------------------------------------------------------------------------
__device__ __forceinline__ void async_load_b128(void* lds_dst, const void* gsrc) {
  unsigned int ldsa = (unsigned int)(size_t)lds_dst;
  asm volatile("global_load_async_to_lds_b128 %0, %1, off"
               :: "v"(ldsa), "v"(gsrc) : "memory");
}
__device__ __forceinline__ void wait_async0() {
  asm volatile("s_wait_asynccnt 0x0" ::: "memory");
}

// ---------------------------------------------------------------------------
// CDNA5 Tensor Data Mover: descriptor-driven 2D DMA global->LDS, tracked by
// TENSORcnt. D# bit-packing per ISA 8.3-8.6:
//  group0: [1:0]=count(1), [63:32]=lds_addr, [120:64]=global_addr, [127:126]=type(2)
//  group1: [17:16]=data_size(1 => 2 bytes), [79:48]=tensor_dim0,
//          [111:80]=tensor_dim1, [127:112]=tile_dim0, [143:128]=tile_dim1,
//          [207:160]=tensor_dim0_stride (elements)
//  groups 2/3: zero (2D tensor; higher dims unused)
// This toolchain exposes the 6-arg builtin:
//   (uint32x4 g0, int32x8 g1, int32x4 g2, int32x4 g3, int32x8 g4, i32 cpol)
// ---------------------------------------------------------------------------
#if defined(__has_builtin)
#if __has_builtin(__builtin_amdgcn_tensor_load_to_lds)
#define HAVE_TDM 1
#endif
#endif

#ifdef HAVE_TDM
__device__ __forceinline__ void tdm_load_2d(void* lds_dst, const void* gsrc,
                                            unsigned int tile_w,   // elements/row
                                            unsigned int tile_h,   // rows
                                            unsigned int stride) { // elements
  unsigned long long ga = (unsigned long long)(size_t)gsrc;
  v4u g0;
  g0[0] = 1u;                                      // count=1, user descriptor
  g0[1] = (unsigned int)(size_t)lds_dst;           // lds_addr (bytes)
  g0[2] = (unsigned int)ga;                        // global_addr[31:0]
  g0[3] = (unsigned int)((ga >> 32) & 0x01FFFFFFu) | (2u << 30);  // addr[56:32]|type=2
  v8i g1;
  g1[0] = (int)(1u << 16);                         // data_size=1 (2B); wg_mask=0
  g1[1] = (int)((stride & 0xFFFFu) << 16);         // tensor_dim0[15:0]
  g1[2] = (int)(((stride >> 16) & 0xFFFFu) | ((tile_h & 0xFFFFu) << 16)); // dim0 hi|dim1 lo
  g1[3] = (int)(((tile_h >> 16) & 0xFFFFu) | (tile_w << 16));             // dim1 hi|tile_dim0
  g1[4] = (int)(tile_h & 0xFFFFu);                 // tile_dim1 | tile_dim2=0
  g1[5] = (int)stride;                             // tensor_dim0_stride[31:0]
  g1[6] = 0;
  g1[7] = 0;
  v4i gz4 = {};
  v8i gz8 = {};
  __builtin_amdgcn_tensor_load_to_lds(g0, g1, gz4, gz4, gz8, 0);
}
#endif

// ---------------------------------------------------------------------------
// Setup kernels
// ---------------------------------------------------------------------------
__global__ __launch_bounds__(256) void init_kernel(f16* hstate, int* cnt) {
  int idx = blockIdx.x * 256 + threadIdx.x;
  if (idx < 2 * 2 * B_ * D_) hstate[idx] = (f16)0.0f;
  if (idx < 2) cnt[idx] = 0;
}

// src [K][N] f32 -> dst [N][K] f16
__global__ __launch_bounds__(256) void transpose_w_kernel(const float* __restrict__ src,
                                                          f16* __restrict__ dst, int K, int N) {
  int idx = blockIdx.x * 256 + threadIdx.x;
  if (idx < K * N) {
    int n = idx / K, k = idx % K;
    dst[idx] = (f16)src[(size_t)k * N + n];
  }
}

__global__ __launch_bounds__(256) void embed_kernel(const int* __restrict__ x,
                                                    const float* __restrict__ embed,
                                                    f16* __restrict__ out) {
  int rowi = blockIdx.x;                 // 0..BT-1  (row = b*T + t)
  int tok  = x[rowi];
  out[(size_t)rowi * D_ + threadIdx.x] = (f16)embed[(size_t)tok * D_ + threadIdx.x];
}

// ---------------------------------------------------------------------------
// WMMA GEMM: C[M,N] = A[M,K](f16) @ Wt[N,K](f16) + bias
// 128 thr = 4 waves; block tile 64 x (NT*16); wave w owns rows w*16..w*16+15.
// Branch-free inner loop (NT is a template arg) + manual 2-stage fragment
// double buffering so loads overlap WMMA (loads retire in order -> partial
// s_wait_loadcnt instead of 0).
// MODE 0: f16 row-major; MODE 1: f32 row-major; MODE 2: f16 -> vT [B][D][T]
// ---------------------------------------------------------------------------
template <int NT, int MODE>
__global__ __launch_bounds__(128) void gemm_wmma_kernel(
    const f16* __restrict__ A, const f16* __restrict__ Wt,
    const float* __restrict__ bias, void* __restrict__ Cout,
    int M, int N, int K) {
  int lane = threadIdx.x & 31;
  int wave = threadIdx.x >> 5;
  int m0 = blockIdx.x * 64 + wave * 16;
  int nbase = blockIdx.y * (NT * 16);
  v8f zero = {};
  v8f acc[NT];
#pragma unroll
  for (int j = 0; j < NT; ++j) acc[j] = zero;

  v16h a0 = load_a_frag(A, K, m0, 0, lane);
  v16h b0[NT], a1, b1[NT];
#pragma unroll
  for (int j = 0; j < NT; ++j) b0[j] = load_b_frag(Wt, K, nbase + j * 16, 0, lane);

  for (int kk = 0; kk < K; kk += 64) {
    // stage A: consume kk, prefetch kk+32
    a1 = load_a_frag(A, K, m0, kk + 32, lane);
#pragma unroll
    for (int j = 0; j < NT; ++j) b1[j] = load_b_frag(Wt, K, nbase + j * 16, kk + 32, lane);
#pragma unroll
    for (int j = 0; j < NT; ++j) acc[j] = wmma_f16(a0, b0[j], acc[j]);
    // stage B: consume kk+32, prefetch kk+64
    if (kk + 64 < K) {
      a0 = load_a_frag(A, K, m0, kk + 64, lane);
#pragma unroll
      for (int j = 0; j < NT; ++j) b0[j] = load_b_frag(Wt, K, nbase + j * 16, kk + 64, lane);
    }
#pragma unroll
    for (int j = 0; j < NT; ++j) acc[j] = wmma_f16(a1, b1[j], acc[j]);
  }

  int rbase = m0 + ((lane & 16) ? 8 : 0);
#pragma unroll
  for (int j = 0; j < NT; ++j) {
    int col = nbase + j * 16 + (lane & 15);
    float bv = bias ? bias[col] : 0.0f;
#pragma unroll
    for (int r = 0; r < 8; ++r) {
      int row = rbase + r;
      float val = acc[j][r] + bv;
      if (MODE == 0) {
        ((f16*)Cout)[(size_t)row * N + col] = (f16)val;
      } else if (MODE == 1) {
        ((float*)Cout)[(size_t)row * N + col] = val;
      } else {                                   // vT: [b][d][t]
        int bb = row >> 9, t = row & (T_ - 1);
        ((f16*)Cout)[((size_t)bb * D_ + col) * T_ + t] = (f16)val;
      }
    }
  }
}

// ---------------------------------------------------------------------------
// Persistent GRU recurrence. grid = 8 blocks: dir = blk>>2 (0=fwd,1=bwd),
// slice = blk&3 owns h-columns [slice*64, slice*64+64).
// whT slice (192x256 f16 = 96KB) is staged once into LDS by the Tensor Data
// Mover (one 2D descriptor per gate slab, TENSORcnt) and stays resident all
// 512 steps (possible only with CDNA5's 320KB LDS). Per step, previous h
// [64x256] f16 is async-staged to LDS (ASYNCcnt), ph = h@wh runs on WMMA,
// gates on VALU, and the 4 slices of a direction sync via an agent-scope
// atomic barrier (s_cluster_barrier issued as the cluster-dispatch fast
// path; NOP when not dispatched as a cluster).
// ---------------------------------------------------------------------------
__global__ __launch_bounds__(256) void gru_kernel(
    const f16* __restrict__ pxf, const f16* __restrict__ pxb,
    const f16* __restrict__ whTf, const f16* __restrict__ whTb,
    const float* __restrict__ bhf, const float* __restrict__ bhb,
    f16* __restrict__ hcat, f16* __restrict__ hstate, int* __restrict__ cnt) {
  __shared__ f16   s_wh[192 * 256];   // 96KB: own whT rows {g*256+c0+j}
  __shared__ f16   s_h[64 * 256];     // 32KB: full previous h
  __shared__ float s_bh[192];

  const int tid  = threadIdx.x;
  const int lane = tid & 31;
  const int wave = tid >> 5;
  const int dir   = blockIdx.x >> 2;
  const int slice = blockIdx.x & 3;
  const int c0 = slice * 64;

  const f16*   px  = dir ? pxb  : pxf;
  const f16*   whT = dir ? whTb : whTf;
  const float* bh  = dir ? bhb  : bhf;
  f16* hs0 = hstate + (size_t)dir * 2 * B_ * D_;

  // one-time: stage recurrent weight slice into LDS.
#ifdef HAVE_TDM
  // TDM: one 2D tile per gate slab (64 rows x 256 f16, row stride 256).
  if (wave == 0) {
    for (int g = 0; g < 3; ++g)
      tdm_load_2d(&s_wh[g * 64 * 256], whT + (size_t)(g * 256 + c0) * 256,
                  256u, 64u, 256u);
    __builtin_amdgcn_s_wait_tensorcnt(0);
  }
#else
  for (int c = tid; c < 192 * 32; c += 256) {          // 32x16B chunks per row
    int rr = c >> 5, ck = c & 31;
    int g = rr >> 6, j = rr & 63;
    async_load_b128(&s_wh[c * 8], whT + (size_t)(g * 256 + c0 + j) * 256 + ck * 8);
  }
  wait_async0();
#endif
  if (tid < 192) {
    int g = tid >> 6, j = tid & 63;
    s_bh[tid] = bh[g * 256 + c0 + j];
  }
  __syncthreads();

  const int mtile = wave & 3;   // rows m0 = mtile*16 of the 64-batch block
  const int cp    = wave >> 2;  // col-tile pair {2cp, 2cp+1} within 64 cols
  const int m0 = mtile * 16;

  for (int step = 0; step < T_; ++step) {
    int t = dir ? (T_ - 1 - step) : step;

    // async-stage previous h (written by all 4 slices last step) -> LDS
    const f16* hin = hs0 + (size_t)(step & 1) * (B_ * D_);
    for (int c = tid; c < (B_ * D_) / 8; c += 256)
      async_load_b128(&s_h[c * 8], hin + c * 8);
    wait_async0();
    __syncthreads();

    // ph = h @ wh (own 3 gates x 2 col-tiles), K=256 in 8 WMMA steps
    v8f zero = {};
    v8f acc[3][2];
#pragma unroll
    for (int g = 0; g < 3; ++g)
#pragma unroll
      for (int ci = 0; ci < 2; ++ci) acc[g][ci] = zero;
#pragma unroll
    for (int kk = 0; kk < 256; kk += 32) {
      v16h a = load_a_frag(s_h, 256, m0, kk, lane);
#pragma unroll
      for (int g = 0; g < 3; ++g)
#pragma unroll
        for (int ci = 0; ci < 2; ++ci) {
          int rr0 = g * 64 + (cp * 2 + ci) * 16;
          v16h b = load_b_frag(s_wh, 256, rr0, kk, lane);
          acc[g][ci] = wmma_f16(a, b, acc[g][ci]);
        }
    }

    // gates + state update
    f16* hout = hs0 + (size_t)((step + 1) & 1) * (B_ * D_);
#pragma unroll
    for (int ci = 0; ci < 2; ++ci) {
      int nloc = (cp * 2 + ci) * 16 + (lane & 15);
      int col  = c0 + nloc;
      float bz = s_bh[nloc], br = s_bh[64 + nloc], bg = s_bh[128 + nloc];
#pragma unroll
      for (int r = 0; r < 8; ++r) {
        int brow = m0 + r + ((lane & 16) ? 8 : 0);
        size_t pxrow = ((size_t)brow * T_ + t) * G_;
        float xz = (float)px[pxrow + col];
        float xr = (float)px[pxrow + 256 + col];
        float xh = (float)px[pxrow + 512 + col];
        float hz = acc[0][ci][r] + bz;
        float hr = acc[1][ci][r] + br;
        float hh = acc[2][ci][r] + bg;
        float z  = 1.0f / (1.0f + __expf(-(xz + hz)));
        float rg = 1.0f / (1.0f + __expf(-(xr + hr)));
        float hc = tanhf(xh + rg * hh);
        float hp = (float)s_h[brow * 256 + col];
        float hn = z * hp + (1.0f - z) * hc;
        hout[brow * 256 + col] = (f16)hn;
        hcat[((size_t)brow * T_ + t) * D2_ + dir * 256 + col] = (f16)hn;
      }
      // speculative prefetch of next step's px row (global_prefetch_b8)
      if (step + 1 < T_) {
        int tn = dir ? (t - 1) : (t + 1);
        __builtin_prefetch(&px[((size_t)(m0 + (lane & 15)) * T_ + tn) * G_ + col], 0, 1);
      }
    }
    __syncthreads();

    // cluster fast path (NOP if not dispatched as a cluster)
    __builtin_amdgcn_s_cluster_barrier();
    // portable inter-WGP barrier among the 4 slices of this direction
    if (tid == 0) {
      __hip_atomic_fetch_add(&cnt[dir], 1, __ATOMIC_ACQ_REL, __HIP_MEMORY_SCOPE_AGENT);
      int target = 4 * (step + 1);
      while (__hip_atomic_load(&cnt[dir], __ATOMIC_ACQUIRE, __HIP_MEMORY_SCOPE_AGENT) < target)
        __builtin_amdgcn_s_sleep(1);
    }
    __syncthreads();
  }
}

// ---------------------------------------------------------------------------
// Fused attention: per (batch, 16-row tile of T). scores -> softmax -> @v.
// Never materializes [B,T,T] in HBM. 128 threads = 4 waves; k-outer loops so
// each A fragment feeds 8 (phase 1) / 4 (phase 3) WMMAs.
// ---------------------------------------------------------------------------
__global__ __launch_bounds__(128) void attention_kernel(
    const f16* __restrict__ q16, const f16* __restrict__ k16,
    const f16* __restrict__ vT16, f16* __restrict__ a16) {
  __shared__ float s_sc[16 * 512];    // 32KB scores
  __shared__ f16   s_w[16 * 512];     // 16KB softmaxed weights
  __shared__ float s_red[16 * 8];

  int tid = threadIdx.x, lane = tid & 31, wave = tid >> 5;
  int b  = blockIdx.x >> 5;
  int t0 = (blockIdx.x & 31) * 16;
  const f16* qb = q16  + (size_t)b * T_ * D_;
  const f16* kb = k16  + (size_t)b * T_ * D_;
  const f16* vb = vT16 + (size_t)b * D_ * T_;   // [D][T] == [N,K] layout

  // phase 1: scores[16 x 512] = q(t0:) @ k^T (k rows are already [N,K])
  {
    v8f acc8[8];
    v8f zero = {};
#pragma unroll
    for (int nt = 0; nt < 8; ++nt) acc8[nt] = zero;
    for (int kk = 0; kk < D_; kk += 32) {
      v16h a = load_a_frag(qb, D_, t0, kk, lane);
#pragma unroll
      for (int nt = 0; nt < 8; ++nt) {
        v16h bf = load_b_frag(kb, D_, (wave * 8 + nt) * 16, kk, lane);
        acc8[nt] = wmma_f16(a, bf, acc8[nt]);
      }
    }
    int rb = (lane & 16) ? 8 : 0;
#pragma unroll
    for (int nt = 0; nt < 8; ++nt) {
      int col = (wave * 8 + nt) * 16 + (lane & 15);
#pragma unroll
      for (int r = 0; r < 8; ++r) s_sc[(rb + r) * 512 + col] = acc8[nt][r];
    }
  }
  __syncthreads();

  // phase 2: softmax over 512 per row; 8 threads per row
  int row = tid >> 3, sub = tid & 7;
  float mx = -1e30f;
  for (int s = sub * 64; s < sub * 64 + 64; ++s) mx = fmaxf(mx, s_sc[row * 512 + s]);
  s_red[row * 8 + sub] = mx;
  __syncthreads();
  float rmx = s_red[row * 8];
  for (int i = 1; i < 8; ++i) rmx = fmaxf(rmx, s_red[row * 8 + i]);
  __syncthreads();
  float sm = 0.f;
  for (int s = sub * 64; s < sub * 64 + 64; ++s) sm += __expf(s_sc[row * 512 + s] - rmx);
  s_red[row * 8 + sub] = sm;
  __syncthreads();
  float rsum = 0.f;
  for (int i = 0; i < 8; ++i) rsum += s_red[row * 8 + i];
  float inv = 1.0f / rsum;
  for (int s = sub * 64; s < sub * 64 + 64; ++s)
    s_w[row * 512 + s] = (f16)(__expf(s_sc[row * 512 + s] - rmx) * inv);
  __syncthreads();

  // phase 3: a[16 x 256] = w @ v  (vT gives contiguous K per output col)
  {
    v8f acc4[4];
    v8f zero = {};
#pragma unroll
    for (int j = 0; j < 4; ++j) acc4[j] = zero;
    for (int kk = 0; kk < T_; kk += 32) {
      v16h a = load_a_frag(s_w, 512, 0, kk, lane);
#pragma unroll
      for (int j = 0; j < 4; ++j) {
        v16h bf = load_b_frag(vb, T_, (wave * 4 + j) * 16, kk, lane);
        acc4[j] = wmma_f16(a, bf, acc4[j]);
      }
    }
    int rb = (lane & 16) ? 8 : 0;
#pragma unroll
    for (int j = 0; j < 4; ++j) {
      int col = (wave * 4 + j) * 16 + (lane & 15);
#pragma unroll
      for (int r = 0; r < 8; ++r)
        a16[((size_t)b * T_ + t0 + rb + r) * D_ + col] = (f16)acc4[j][r];
    }
  }
}

// ---------------------------------------------------------------------------
// CRF: forward (log-likelihood) + Viterbi + backtrack. One block per batch,
// thread j owns state j (N=48). trans cached transposed in LDS.
// ---------------------------------------------------------------------------
__global__ __launch_bounds__(64) void crf_kernel(
    const float* __restrict__ logits, const int* __restrict__ y,
    const int* __restrict__ seq_len, const float* __restrict__ trans,
    int* __restrict__ bp, float* __restrict__ out) {
  __shared__ float s_tT[48 * 49];
  __shared__ float s_all[48], s_alv[48];
  __shared__ float s_red[64];

  int b = blockIdx.x, tid = threadIdx.x;
  int L = seq_len[b];
  const float* lg = logits + (size_t)b * T_ * N_;
  const int* yb = y + (size_t)b * T_;

  for (int i = tid; i < 48 * 48; i += 64) {
    int j = i / 48, ii = i % 48;
    s_tT[j * 49 + ii] = trans[ii * 48 + j];
  }
  if (tid < 48) { float e0 = lg[tid]; s_all[tid] = e0; s_alv[tid] = e0; }
  float us = 0.f;
  for (int t = tid; t < T_; t += 64) {
    if (t < L) us += lg[t * N_ + yb[t]];
    if (t >= 1 && t < L) us += trans[yb[t - 1] * N_ + yb[t]];
  }
  s_red[tid] = us;
  __syncthreads();

  for (int t = 1; t < T_; ++t) {
    int mt = (t < L);
    float nll = 0.f, nlv = 0.f;
    if (tid < 48) {
      float e = lg[t * N_ + tid];
      float m1 = -1e30f, m2 = -1e30f; int am = 0;
      for (int i = 0; i < 48; ++i) {
        float tr = s_tT[tid * 49 + i];
        float s1 = s_all[i] + tr;  if (s1 > m1) m1 = s1;
        float s2 = s_alv[i] + tr;  if (s2 > m2) { m2 = s2; am = i; }
      }
      float accs = 0.f;
      for (int i = 0; i < 48; ++i) accs += __expf(s_all[i] + s_tT[tid * 49 + i] - m1);
      nll = mt ? (m1 + __logf(accs) + e) : s_all[tid];
      nlv = mt ? (m2 + e) : s_alv[tid];
      bp[((size_t)b * T_ + t) * N_ + tid] = mt ? am : tid;
    }
    __syncthreads();
    if (tid < 48) { s_all[tid] = nll; s_alv[tid] = nlv; }
    __syncthreads();
  }

  if (tid == 0) {
    float score = 0.f;
    for (int i = 0; i < 64; ++i) score += s_red[i];
    float m = -1e30f;
    for (int i = 0; i < 48; ++i) m = fmaxf(m, s_all[i]);
    float a2 = 0.f;
    for (int i = 0; i < 48; ++i) a2 += __expf(s_all[i] - m);
    out[BT_ + b] = score - (m + __logf(a2));

    float bm = -1e30f; int tag = 0;
    for (int i = 0; i < 48; ++i) if (s_alv[i] > bm) { bm = s_alv[i]; tag = i; }
    for (int t = T_ - 1; t >= 1; --t) {
      out[(size_t)b * T_ + t] = (float)tag;
      tag = bp[((size_t)b * T_ + t) * N_ + tag];
    }
    out[(size_t)b * T_] = (float)tag;
  }
}

// ---------------------------------------------------------------------------
extern "C" void kernel_launch(void* const* d_in, const int* in_sizes, int n_in,
                              void* d_out, int out_size, void* d_ws, size_t ws_size,
                              hipStream_t stream) {
  const int*   x      = (const int*)d_in[0];
  const int*   y      = (const int*)d_in[1];
  const int*   seqlen = (const int*)d_in[2];
  const float* embed  = (const float*)d_in[3];
  const float* wx_f   = (const float*)d_in[4];
  const float* wh_f   = (const float*)d_in[5];
  const float* b_f    = (const float*)d_in[6];
  const float* wx_b   = (const float*)d_in[7];
  const float* wh_b   = (const float*)d_in[8];
  const float* b_b    = (const float*)d_in[9];
  const float* wq     = (const float*)d_in[10];
  const float* bq     = (const float*)d_in[11];
  const float* wk     = (const float*)d_in[12];
  const float* bk     = (const float*)d_in[13];
  const float* wv     = (const float*)d_in[14];
  const float* bv     = (const float*)d_in[15];
  const float* wt     = (const float*)d_in[16];
  const float* bt     = (const float*)d_in[17];
  const float* trans  = (const float*)d_in[18];
  float* out = (float*)d_out;
  (void)in_sizes; (void)n_in; (void)out_size; (void)ws_size;

  char* ws = (char*)d_ws;
  size_t off = 0;
  auto alloc = [&](size_t bytes) { char* p = ws + off; off += (bytes + 255) & ~(size_t)255; return p; };

  f16* emb16  = (f16*)alloc((size_t)BT_ * D_ * 2);
  f16* wxTf   = (f16*)alloc((size_t)G_ * D_ * 2);
  f16* wxTb   = (f16*)alloc((size_t)G_ * D_ * 2);
  f16* whTf   = (f16*)alloc((size_t)G_ * D_ * 2);
  f16* whTb   = (f16*)alloc((size_t)G_ * D_ * 2);
  f16* wqT    = (f16*)alloc((size_t)D_ * D2_ * 2);
  f16* wkT    = (f16*)alloc((size_t)D_ * D2_ * 2);
  f16* wvT    = (f16*)alloc((size_t)D_ * D2_ * 2);
  f16* wtT    = (f16*)alloc((size_t)N_ * D_ * 2);
  f16* px16f  = (f16*)alloc((size_t)BT_ * G_ * 2);
  f16* px16b  = (f16*)alloc((size_t)BT_ * G_ * 2);
  f16* hcat16 = (f16*)alloc((size_t)BT_ * D2_ * 2);
  f16* q16    = (f16*)alloc((size_t)BT_ * D_ * 2);
  f16* k16    = (f16*)alloc((size_t)BT_ * D_ * 2);
  f16* vT16   = (f16*)alloc((size_t)BT_ * D_ * 2);
  f16* a16    = (f16*)alloc((size_t)BT_ * D_ * 2);
  float* logits = (float*)alloc((size_t)BT_ * N_ * 4);
  int*  bpbuf = (int*)alloc((size_t)BT_ * N_ * 4);
  f16* hstate = (f16*)alloc((size_t)2 * 2 * B_ * D_ * 2);
  int* cnt    = (int*)alloc(256);

  init_kernel<<<256, 256, 0, stream>>>(hstate, cnt);

  transpose_w_kernel<<<(D_ * G_ + 255) / 256, 256, 0, stream>>>(wx_f, wxTf, D_, G_);
  transpose_w_kernel<<<(D_ * G_ + 255) / 256, 256, 0, stream>>>(wh_f, whTf, D_, G_);
  transpose_w_kernel<<<(D_ * G_ + 255) / 256, 256, 0, stream>>>(wx_b, wxTb, D_, G_);
  transpose_w_kernel<<<(D_ * G_ + 255) / 256, 256, 0, stream>>>(wh_b, whTb, D_, G_);
  transpose_w_kernel<<<(D2_ * D_ + 255) / 256, 256, 0, stream>>>(wq, wqT, D2_, D_);
  transpose_w_kernel<<<(D2_ * D_ + 255) / 256, 256, 0, stream>>>(wk, wkT, D2_, D_);
  transpose_w_kernel<<<(D2_ * D_ + 255) / 256, 256, 0, stream>>>(wv, wvT, D2_, D_);
  transpose_w_kernel<<<(D_ * N_ + 255) / 256, 256, 0, stream>>>(wt, wtT, D_, N_);

  embed_kernel<<<BT_, 256, 0, stream>>>(x, embed, emb16);

  // px = emb @ wx + bx  (hoisted out of the scan; fully parallel over B*T)
  gemm_wmma_kernel<4, 0><<<dim3(BT_ / 64, G_ / 64), 128, 0, stream>>>(emb16, wxTf, b_f, px16f, BT_, G_, D_);
  gemm_wmma_kernel<4, 0><<<dim3(BT_ / 64, G_ / 64), 128, 0, stream>>>(emb16, wxTb, b_b, px16b, BT_, G_, D_);

  // sequential recurrence: 2 dirs x 4 WGP slices, LDS-resident wh (TDM-staged)
  gru_kernel<<<8, 256, 0, stream>>>(px16f, px16b, whTf, whTb,
                                    b_f + G_, b_b + G_, hcat16, hstate, cnt);

  // q/k/v projections from concatenated h
  gemm_wmma_kernel<4, 0><<<dim3(BT_ / 64, D_ / 64), 128, 0, stream>>>(hcat16, wqT, bq, q16, BT_, D_, D2_);
  gemm_wmma_kernel<4, 0><<<dim3(BT_ / 64, D_ / 64), 128, 0, stream>>>(hcat16, wkT, bk, k16, BT_, D_, D2_);
  gemm_wmma_kernel<4, 2><<<dim3(BT_ / 64, D_ / 64), 128, 0, stream>>>(hcat16, wvT, bv, vT16, BT_, D_, D2_);

  attention_kernel<<<B_ * (T_ / 16), 128, 0, stream>>>(q16, k16, vT16, a16);

  // logits: N=48 -> 3 full 16-wide tiles, branch-free
  gemm_wmma_kernel<3, 1><<<dim3(BT_ / 64, 1), 128, 0, stream>>>(a16, wtT, bt, logits, BT_, N_, D_);

  crf_kernel<<<B_, 64, 0, stream>>>(logits, y, seqlen, trans, bpbuf, out);
}
